// GPTTNBlock_23063974380101
// MI455X (gfx1250) — matrix operand order repaired and checked by reading, not compiled, not run on hardware
//
#include <hip/hip_runtime.h>
#include <hip/hip_bf16.h>
#include <cstdint>
#include <cstddef>

typedef __bf16 bf16;
typedef __attribute__((ext_vector_type(16))) __bf16 v16bf;
typedef __attribute__((ext_vector_type(8)))  float  v8f;

#define D_MODEL 1024
#define NHEADS  8
#define DH      32
#define DFF     4096
#define BATCH   4
#define SEQ     2048
#define BTOK    (BATCH*SEQ)          // 8192 token rows
#define NA      (NHEADS*DH*DH)       // 8192 columns of Wa
#define HDIM    (NHEADS*DH)          // 256
#define SCHUNK  256
#define LG_SCHUNK 8
#define NCHUNK  (SEQ/SCHUNK)

#define LDSSTR  40                   // bf16 elems per LDS row (80B: 16B-aligned, bank-spread)
#define BUFELT  (128*LDSSTR)         // elements per tile buffer (5120)
#define BUFBYTE (BUFELT*2)           // 10240 bytes

// ---------------------------------------------------------------------------
// CDNA5 async global->LDS copy (16B per lane), tracked by ASYNCcnt.
// ---------------------------------------------------------------------------
__device__ __forceinline__ void async_b128(uint32_t lds_off, const bf16* gptr)
{
    asm volatile("global_load_async_to_lds_b128 %0, %1, off"
                 :: "v"(lds_off), "v"((uint64_t)(uintptr_t)gptr)
                 : "memory");
}

// ---------------------------------------------------------------------------
// Weight convert + transpose: in[K][N] f32  ->  out[N][K] bf16
// ---------------------------------------------------------------------------
__global__ __launch_bounds__(256)
void transpose_f32_bf16(const float* __restrict__ in, bf16* __restrict__ out,
                        int K, int N)
{
    __shared__ float tile[32][33];
    const int tx = threadIdx.x, ty = threadIdx.y;
    const int n0 = blockIdx.x * 32, k0 = blockIdx.y * 32;
#pragma unroll
    for (int r = 0; r < 4; ++r)
        tile[ty + r*8][tx] = in[(size_t)(k0 + ty + r*8) * N + n0 + tx];
    __syncthreads();
#pragma unroll
    for (int r = 0; r < 4; ++r)
        out[(size_t)(n0 + ty + r*8) * K + k0 + tx] = (bf16)tile[tx][ty + r*8];
}

// ---------------------------------------------------------------------------
// LayerNorm (fp32 in) -> bf16 out.  One row (D=1024) per 256-thread block.
// ---------------------------------------------------------------------------
__global__ __launch_bounds__(256)
void layernorm_bf16(const float* __restrict__ x, const float* __restrict__ g,
                    const float* __restrict__ bta, bf16* __restrict__ out)
{
    const int row = blockIdx.x;
    const int t   = threadIdx.x;
    const float* xr = x + (size_t)row * D_MODEL;
    float v[4], s = 0.f, ss = 0.f;
#pragma unroll
    for (int i = 0; i < 4; ++i) { v[i] = xr[t + i*256]; s += v[i]; ss += v[i]*v[i]; }
#pragma unroll
    for (int off = 16; off >= 1; off >>= 1) {
        s  += __shfl_xor(s,  off, 32);
        ss += __shfl_xor(ss, off, 32);
    }
    __shared__ float ps[8], pss[8];
    if ((t & 31) == 0) { ps[t >> 5] = s; pss[t >> 5] = ss; }
    __syncthreads();
    s = 0.f; ss = 0.f;
#pragma unroll
    for (int w = 0; w < 8; ++w) { s += ps[w]; ss += pss[w]; }
    const float mean = s  * (1.f / D_MODEL);
    const float var  = ss * (1.f / D_MODEL) - mean * mean;
    const float inv  = rsqrtf(var + 1e-5f);
#pragma unroll
    for (int i = 0; i < 4; ++i) {
        int c = t + i*256;
        out[(size_t)row * D_MODEL + c] = (bf16)((v[i] - mean) * inv * g[c] + bta[c]);
    }
}

// ---------------------------------------------------------------------------
// bf16 WMMA GEMM:  C[M,N] = A[M,K] * Bt[N,K]^T  (+ epilogue)
//   EPI 0: out f32 = acc + bias[n]
//   EPI 1: out f32 = acc + bias[n] + resid[m,n]
//   EPI 2: out bf16 = gelu_exact(acc + bias[n])
// Block tile 128x128, K-step 32, 8 waves (4x2 wave grid, 2x4 wmma tiles each).
// Global->LDS staging uses CDNA5 async-to-LDS copies with double-buffered LDS:
// tile i+1 streams in (ASYNCcnt) while tile i feeds the WMMAs.
// mapA!=0 remaps A rows (chunked token slabs): mem_row = (r>>8)*stot + cstart + (r&255)
// ---------------------------------------------------------------------------
union Frag { v16bf v; uint4 q[2]; };

__device__ __forceinline__ size_t map_row(int r, int mapA, int stot, int cstart)
{
    if (!mapA) return (size_t)r;
    return (size_t)(r >> LG_SCHUNK) * stot + cstart + (r & (SCHUNK - 1));
}

template<int EPI>
__global__ __launch_bounds__(256)
void gemm_bf16(const bf16* __restrict__ A, const bf16* __restrict__ Bt,
               const float* __restrict__ bias, const float* __restrict__ resid,
               void* __restrict__ outp, int M, int N, int K,
               int mapA, int stot, int cstart)
{
    __shared__ __align__(16) bf16 As[2 * BUFELT];   // double-buffered A tile
    __shared__ __align__(16) bf16 Bs[2 * BUFELT];   // double-buffered B tile

    const int t  = threadIdx.x;
    const int bm = blockIdx.y, bn = blockIdx.x;

    // global->LDS staging map: each thread owns one 16B chunk in rows r and r+64
    const int ldRow = t >> 2;            // 0..63
    const int ldOff = (t & 3) << 3;      // bf16 offset 0/8/16/24
    const bf16* gA0 = A  + map_row(bm*128 + ldRow,      mapA, stot, cstart) * K + ldOff;
    const bf16* gA1 = A  + map_row(bm*128 + ldRow + 64, mapA, stot, cstart) * K + ldOff;
    const bf16* gB0 = Bt + (size_t)(bn*128 + ldRow)      * K + ldOff;
    const bf16* gB1 = Bt + (size_t)(bn*128 + ldRow + 64) * K + ldOff;

    // LDS byte offsets of this thread's two chunks in each tile (per buffer)
    const uint32_t aBase = (uint32_t)(uintptr_t)As;
    const uint32_t bBase = (uint32_t)(uintptr_t)Bs;
    const uint32_t cOff0 = (uint32_t)(ldRow * LDSSTR + ldOff) * 2u;
    const uint32_t cOff1 = (uint32_t)((ldRow + 64) * LDSSTR + ldOff) * 2u;

    const int wave  = t >> 5;
    const int lane  = t & 31;
    const int half  = lane >> 4;         // 0/1 half-wave
    const int l15   = lane & 15;
    const int mWave = (wave & 3) * 32;   // 4 M-groups of 32 rows
    const int nWave = (wave >> 2) * 64;  // 2 N-groups of 64 cols

    v8f acc[2][4];
#pragma unroll
    for (int mi = 0; mi < 2; ++mi)
#pragma unroll
        for (int ni = 0; ni < 4; ++ni)
#pragma unroll
            for (int e = 0; e < 8; ++e) acc[mi][ni][e] = 0.f;

    // prologue: stream tile 0 into buffer 0
    async_b128(aBase + cOff0, gA0);
    async_b128(aBase + cOff1, gA1);
    async_b128(bBase + cOff0, gB0);
    async_b128(bBase + cOff1, gB1);

    int cur = 0;
    for (int k0 = 0; k0 < K; k0 += 32) {
        const int nxt = cur ^ 1;
        if (k0 + 32 < K) {               // stream next tile while this one computes
            const int kn = k0 + 32;
            const uint32_t ab = aBase + (uint32_t)nxt * BUFBYTE;
            const uint32_t bb = bBase + (uint32_t)nxt * BUFBYTE;
            async_b128(ab + cOff0, gA0 + kn);
            async_b128(ab + cOff1, gA1 + kn);
            async_b128(bb + cOff0, gB0 + kn);
            async_b128(bb + cOff1, gB1 + kn);
            // 4 newest remain outstanding; the 4 for `cur` are done (in-order)
            asm volatile("s_wait_asynccnt 0x4" ::: "memory");
        } else {
            asm volatile("s_wait_asynccnt 0x0" ::: "memory");
        }
        __syncthreads();                 // whole tile valid across all waves

        const bf16* Asb = As + (size_t)cur * BUFELT;
        const bf16* Bsb = Bs + (size_t)cur * BUFELT;
        Frag fa[2], fb[4];
#pragma unroll
        for (int mi = 0; mi < 2; ++mi) { // A frag: K 0-7/16-23 (half0), 8-15/24-31 (half1)
            int row = mWave + mi*16 + l15;
            fa[mi].q[0] = *(const uint4*)&Asb[row*LDSSTR + half*8];
            fa[mi].q[1] = *(const uint4*)&Asb[row*LDSSTR + half*8 + 16];
        }
#pragma unroll
        for (int ni = 0; ni < 4; ++ni) { // B frag: K 0-7/16-23 (half0), 8-15/24-31 (half1)
            int row = nWave + ni*16 + l15;
            fb[ni].q[0] = *(const uint4*)&Bsb[row*LDSSTR + half*8];
            fb[ni].q[1] = *(const uint4*)&Bsb[row*LDSSTR + half*8 + 16];
        }
#pragma unroll
        for (int mi = 0; mi < 2; ++mi)
#pragma unroll
            for (int ni = 0; ni < 4; ++ni)
                acc[mi][ni] = __builtin_amdgcn_wmma_f32_16x16x32_bf16(
                    false, fa[mi].v, false, fb[ni].v, (short)0, acc[mi][ni],
                    false, false);
        __syncthreads();                 // all reads of `cur` done before it is refilled
        cur = nxt;
    }

    float* outf = (float*)outp;
    bf16*  outh = (bf16*)outp;
#pragma unroll
    for (int mi = 0; mi < 2; ++mi) {
        const int mBase = bm*128 + mWave + mi*16 + half*8;   // C layout: vgpr e -> M=e(+8)
#pragma unroll
        for (int ni = 0; ni < 4; ++ni) {
            const int n  = bn*128 + nWave + ni*16 + l15;     // N = lane
            const float bn_ = bias[n];
#pragma unroll
            for (int e = 0; e < 8; ++e) {
                const int m = mBase + e;
                float val = acc[mi][ni][e] + bn_;
                const size_t idx = (size_t)m * N + n;
                if (EPI == 0) {
                    outf[idx] = val;
                } else if (EPI == 1) {
                    outf[idx] = val + resid[idx];
                } else {
                    float ge = 0.5f * val * (1.f + erff(val * 0.70710678118654752f));
                    outh[idx] = (bf16)ge;
                }
            }
        }
    }
}

// ---------------------------------------------------------------------------
// MPS recurrence over one sequence chunk. One wave32 per (batch,head) chain.
// Lane i owns output row i of the 32x32 transfer matrix; h_j broadcast by shfl.
// ---------------------------------------------------------------------------
__global__ __launch_bounds__(32)
void recurrence_kernel(const float* __restrict__ Achunk,
                       const float* __restrict__ h0,
                       float* __restrict__ hstate,
                       bf16* __restrict__ ys,
                       int Sc, int chunkStart, int first)
{
    const int chain = blockIdx.x;          // 0..31  (b*8 + h)
    const int b  = chain >> 3;
    const int hh = chain & 7;
    const int lane = threadIdx.x;

    float hv = first ? h0[hh*32 + lane] : hstate[chain*32 + lane];

    const float* base = Achunk + (size_t)(b * Sc) * NA + hh*1024 + lane*32;
    float4 buf[8];
#pragma unroll
    for (int q = 0; q < 8; ++q) buf[q] = ((const float4*)base)[q];

    for (int tl = 0; tl < Sc; ++tl) {
        float a[32];
#pragma unroll
        for (int q = 0; q < 8; ++q) {
            a[q*4+0] = buf[q].x; a[q*4+1] = buf[q].y;
            a[q*4+2] = buf[q].z; a[q*4+3] = buf[q].w;
        }
        if (tl + 1 < Sc) {                 // software-pipelined row prefetch
            const float* nxt = base + (size_t)(tl + 1) * NA;
#pragma unroll
            for (int q = 0; q < 8; ++q) buf[q] = ((const float4*)nxt)[q];
        }
        float acc0 = 0.f, acc1 = 0.f, acc2 = 0.f, acc3 = 0.f;
#pragma unroll
        for (int j = 0; j < 32; j += 4) {
            acc0 = fmaf(a[j+0], __shfl(hv, j+0, 32), acc0);
            acc1 = fmaf(a[j+1], __shfl(hv, j+1, 32), acc1);
            acc2 = fmaf(a[j+2], __shfl(hv, j+2, 32), acc2);
            acc3 = fmaf(a[j+3], __shfl(hv, j+3, 32), acc3);
        }
        float acc = (acc0 + acc1) + (acc2 + acc3);
        float s = acc * acc;
#pragma unroll
        for (int off = 16; off >= 1; off >>= 1) s += __shfl_xor(s, off, 32);
        hv = acc / (sqrtf(s) + 1e-6f);
        const int tg = chunkStart + tl;
        ys[(size_t)(b * SEQ + tg) * HDIM + hh*DH + lane] = (bf16)hv;
    }
    hstate[chain*32 + lane] = hv;
}

// ---------------------------------------------------------------------------
extern "C" void kernel_launch(void* const* d_in, const int* in_sizes, int n_in,
                              void* d_out, int out_size, void* d_ws, size_t ws_size,
                              hipStream_t stream)
{
    (void)in_sizes; (void)n_in; (void)out_size; (void)ws_size;
    const float* x    = (const float*)d_in[0];
    const float* ln1g = (const float*)d_in[1];
    const float* ln1b = (const float*)d_in[2];
    const float* Wa   = (const float*)d_in[3];
    const float* ba   = (const float*)d_in[4];
    const float* h0   = (const float*)d_in[5];
    const float* Wo   = (const float*)d_in[6];
    const float* bo   = (const float*)d_in[7];
    const float* ln2g = (const float*)d_in[8];
    const float* ln2b = (const float*)d_in[9];
    const float* W1   = (const float*)d_in[10];
    const float* b1   = (const float*)d_in[11];
    const float* W2   = (const float*)d_in[12];
    const float* b2   = (const float*)d_in[13];

    char* ws = (char*)d_ws;
    size_t off = 0;
    auto wsalloc = [&](size_t bytes) -> void* {
        void* p = ws + off;
        off += (bytes + 255) & ~(size_t)255;
        return p;
    };
    bf16*  WaT    = (bf16*) wsalloc((size_t)NA * D_MODEL * 2);       // [8192,1024]
    bf16*  WoT    = (bf16*) wsalloc((size_t)D_MODEL * HDIM * 2);     // [1024,256]
    bf16*  W1T    = (bf16*) wsalloc((size_t)DFF * D_MODEL * 2);      // [4096,1024]
    bf16*  W2T    = (bf16*) wsalloc((size_t)D_MODEL * DFF * 2);      // [1024,4096]
    bf16*  xln    = (bf16*) wsalloc((size_t)BTOK * D_MODEL * 2);
    float* Achunk = (float*)wsalloc((size_t)BATCH * SCHUNK * NA * 4);// 32MB, reused/chunk
    bf16*  ysb    = (bf16*) wsalloc((size_t)BTOK * HDIM * 2);
    float* hstate = (float*)wsalloc(32 * 32 * 4);
    float* x2     = (float*)wsalloc((size_t)BTOK * D_MODEL * 4);
    bf16*  h2     = (bf16*) wsalloc((size_t)BTOK * D_MODEL * 2);
    bf16*  gact   = (bf16*) wsalloc((size_t)BTOK * DFF * 2);

    const dim3 tb(32, 8);
    transpose_f32_bf16<<<dim3(NA/32,      D_MODEL/32), tb, 0, stream>>>(Wa, WaT, D_MODEL, NA);
    transpose_f32_bf16<<<dim3(D_MODEL/32, HDIM/32),    tb, 0, stream>>>(Wo, WoT, HDIM, D_MODEL);
    transpose_f32_bf16<<<dim3(DFF/32,     D_MODEL/32), tb, 0, stream>>>(W1, W1T, D_MODEL, DFF);
    transpose_f32_bf16<<<dim3(D_MODEL/32, DFF/32),     tb, 0, stream>>>(W2, W2T, DFF, D_MODEL);

    layernorm_bf16<<<BTOK, 256, 0, stream>>>(x, ln1g, ln1b, xln);

    for (int c = 0; c < NCHUNK; ++c) {
        const int Mc = BATCH * SCHUNK;   // 1024 token rows per chunk
        gemm_bf16<0><<<dim3(NA/128, Mc/128), 256, 0, stream>>>(
            xln, WaT, ba, nullptr, Achunk, Mc, NA, D_MODEL,
            /*mapA=*/1, /*stot=*/SEQ, /*cstart=*/c * SCHUNK);
        recurrence_kernel<<<32, 32, 0, stream>>>(
            Achunk, h0, hstate, ysb, SCHUNK, c * SCHUNK, c == 0 ? 1 : 0);
    }

    // x2 = x + ys @ Wo + bo
    gemm_bf16<1><<<dim3(D_MODEL/128, BTOK/128), 256, 0, stream>>>(
        ysb, WoT, bo, x, x2, BTOK, D_MODEL, HDIM, 0, 0, 0);

    layernorm_bf16<<<BTOK, 256, 0, stream>>>(x2, ln2g, ln2b, h2);

    // gact = gelu(h2 @ W1 + b1)   (bf16)
    gemm_bf16<2><<<dim3(DFF/128, BTOK/128), 256, 0, stream>>>(
        h2, W1T, b1, nullptr, gact, BTOK, DFF, D_MODEL, 0, 0, 0);

    // out = x2 + gact @ W2 + b2
    gemm_bf16<1><<<dim3(D_MODEL/128, BTOK/128), 256, 0, stream>>>(
        gact, W2T, b2, x2, d_out, BTOK, D_MODEL, DFF, 0, 0, 0);
}
